// ConvLayer_10488310137184
// MI455X (gfx1250) — matrix operand order repaired and checked by reading, not compiled, not run on hardware
//
#include <hip/hip_runtime.h>
#include <hip/hip_bf16.h>

typedef _Float16 v16h __attribute__((ext_vector_type(16)));
typedef float    v8f  __attribute__((ext_vector_type(8)));

#define N_USERS 50000
#define N_ITEMS 20000
#define N_NODES 70000
#define N_EDGES 320000
#define IN_DIM  256
#define OUT_DIM 256
#define SCALE   0.0625f  // 1/sqrt(256)

// ---------------- zero output ----------------
__global__ void k_zero(float* __restrict__ p, int n4) {
    int i = blockIdx.x * blockDim.x + threadIdx.x;
    if (i < n4) ((float4*)p)[i] = make_float4(0.f, 0.f, 0.f, 0.f);
}

// ---------------- edge attention logits: one wave32 per edge ----------------
__global__ void k_alpha(const float* __restrict__ feat,
                        const int* __restrict__ user_ids,
                        const int* __restrict__ item_ids,
                        const int* __restrict__ edge_src,
                        const int* __restrict__ edge_dst,
                        float* __restrict__ alpha) {
    const int wave = threadIdx.x >> 5;
    const int lane = threadIdx.x & 31;
    const int e = blockIdx.x * (blockDim.x >> 5) + wave;
    if (e >= N_EDGES) return;
    const int sn = user_ids[edge_src[e]];
    const int dn = item_ids[edge_dst[e]];
    const float4* a = (const float4*)(feat + (size_t)sn * IN_DIM) + lane * 2;
    const float4* b = (const float4*)(feat + (size_t)dn * IN_DIM) + lane * 2;
    float4 a0 = a[0], a1 = a[1], b0 = b[0], b1 = b[1];
    float d = a0.x*b0.x + a0.y*b0.y + a0.z*b0.z + a0.w*b0.w
            + a1.x*b1.x + a1.y*b1.y + a1.z*b1.z + a1.w*b1.w;
    #pragma unroll
    for (int off = 16; off; off >>= 1) d += __shfl_down(d, off, 32);
    if (lane == 0) alpha[e] = d * SCALE;
}

// ---------------- deterministic global max (single block) ----------------
__global__ void k_max(const float* __restrict__ alpha, float* __restrict__ scal) {
    __shared__ float red[1024];
    float m = -3.0e38f;
    for (int i = threadIdx.x; i < N_EDGES; i += 1024) m = fmaxf(m, alpha[i]);
    red[threadIdx.x] = m;
    __syncthreads();
    for (int s = 512; s; s >>= 1) {
        if (threadIdx.x < (unsigned)s) red[threadIdx.x] = fmaxf(red[threadIdx.x], red[threadIdx.x + s]);
        __syncthreads();
    }
    if (threadIdx.x == 0) scal[0] = red[0];
}

// ---------------- exp + deterministic sum -> inverse (single block) ----------------
__global__ void k_expsum(const float* __restrict__ alpha, float* __restrict__ ex,
                         float* __restrict__ scal) {
    __shared__ float red[1024];
    const float mv = scal[0];
    float s = 0.f;
    for (int i = threadIdx.x; i < N_EDGES; i += 1024) {
        float v = __expf(alpha[i] - mv);
        ex[i] = v;
        s += v;
    }
    red[threadIdx.x] = s;
    __syncthreads();
    for (int st = 512; st; st >>= 1) {
        if (threadIdx.x < (unsigned)st) red[threadIdx.x] += red[threadIdx.x + st];
        __syncthreads();
    }
    if (threadIdx.x == 0) scal[1] = 1.0f / red[0];
}

// ---------------- fused gather + GEMM (h @ W^T) + bias + relu via WMMA ----------------
// out[m, o] = relu( sum_k feat[ids[m], k] * W[o, k] + bias[o] )
// One wave computes a 16x16 tile; K looped in steps of 32 using v_wmma_f32_16x16x32_f16.
__global__ void k_gemm_relu(const float* __restrict__ feat,
                            const int*   __restrict__ ids,
                            const float* __restrict__ W,
                            const float* __restrict__ bias,
                            float* __restrict__ out, int M) {
    const int wave  = threadIdx.x >> 5;            // 0..3 (128-thread block = 4 waves)
    const int lane  = threadIdx.x & 31;
    const int half  = lane >> 4;                   // 0 | 1
    const int l16   = lane & 15;
    const int mtile = blockIdx.x * 4 + wave;
    const int ntile = blockIdx.y;                  // 0..15
    if (mtile * 16 >= M) return;                   // wave-uniform exit; EXEC stays all-1s

    const int   row  = mtile * 16 + l16;
    const int   node = ids[row];
    const float* arow = feat + (size_t)node * IN_DIM;
    const int   col  = ntile * 16 + l16;
    const float* wrow = W + (size_t)col * IN_DIM;

    v8f acc = {};
    #pragma unroll
    for (int k0 = 0; k0 < IN_DIM; k0 += 32) {
        // A fragment (16x32 f16): lane holds row M=l16, 16 contiguous K at k0 + half*16.
        const float* ap0 = arow + k0 + half * 16;
        const float* ap1 = arow + k0 + half * 16 + 8;
        // B fragment (32x16 f16): lane holds column N=l16, 16 contiguous K at k0 + half*16.
        const float* bp  = wrow + k0 + half * 16;

        float4 a0 = *(const float4*)(ap0);
        float4 a1 = *(const float4*)(ap0 + 4);
        float4 a2 = *(const float4*)(ap1);
        float4 a3 = *(const float4*)(ap1 + 4);
        float4 b0 = *(const float4*)(bp);
        float4 b1 = *(const float4*)(bp + 4);
        float4 b2 = *(const float4*)(bp + 8);
        float4 b3 = *(const float4*)(bp + 12);

        v16h A, B;
        A[0]=(_Float16)a0.x; A[1]=(_Float16)a0.y; A[2]=(_Float16)a0.z; A[3]=(_Float16)a0.w;
        A[4]=(_Float16)a1.x; A[5]=(_Float16)a1.y; A[6]=(_Float16)a1.z; A[7]=(_Float16)a1.w;
        A[8]=(_Float16)a2.x; A[9]=(_Float16)a2.y; A[10]=(_Float16)a2.z; A[11]=(_Float16)a2.w;
        A[12]=(_Float16)a3.x; A[13]=(_Float16)a3.y; A[14]=(_Float16)a3.z; A[15]=(_Float16)a3.w;
        B[0]=(_Float16)b0.x; B[1]=(_Float16)b0.y; B[2]=(_Float16)b0.z; B[3]=(_Float16)b0.w;
        B[4]=(_Float16)b1.x; B[5]=(_Float16)b1.y; B[6]=(_Float16)b1.z; B[7]=(_Float16)b1.w;
        B[8]=(_Float16)b2.x; B[9]=(_Float16)b2.y; B[10]=(_Float16)b2.z; B[11]=(_Float16)b2.w;
        B[12]=(_Float16)b3.x; B[13]=(_Float16)b3.y; B[14]=(_Float16)b3.z; B[15]=(_Float16)b3.w;

        acc = __builtin_amdgcn_wmma_f32_16x16x32_f16(
                  /*neg_a=*/false, A, /*neg_b=*/false, B,
                  /*c_mod=*/(short)0, acc, /*reuse_a=*/false, /*reuse_b=*/false);
    }

    // C/D layout: VGPR r -> M = r + half*8 (lanes 0-15: M=r, lanes 16-31: M=r+8), N = l16
    const float b = bias[col];
    #pragma unroll
    for (int r = 0; r < 8; ++r) {
        const int orow = mtile * 16 + half * 8 + r;
        float v = acc[r] + b;
        out[(size_t)orow * OUT_DIM + col] = v > 0.f ? v : 0.f;
    }
}

// ---------------- edge scatter: both directions, f32 atomics into L2 ----------------
__global__ void k_scatter(const float* __restrict__ feat_src,
                          const float* __restrict__ feat_dst,
                          const int* __restrict__ edge_src,
                          const int* __restrict__ edge_dst,
                          const float* __restrict__ ex,
                          const float* __restrict__ scal,
                          float* __restrict__ out) {
    const int e = blockIdx.x;
    const int d = threadIdx.x;            // 0..255
    const float w  = ex[e] * scal[1];
    const int   u  = edge_src[e];
    const int   it = edge_dst[e];
    // item_new[it] += feat_src[u] * w   (second half of output)
    atomicAdd(out + ((size_t)N_USERS + it) * OUT_DIM + d,
              feat_src[(size_t)u * OUT_DIM + d] * w);
    // user_new[u] += feat_dst[it] * w   (first half of output)
    atomicAdd(out + (size_t)u * OUT_DIM + d,
              feat_dst[(size_t)it * OUT_DIM + d] * w);
}

extern "C" void kernel_launch(void* const* d_in, const int* in_sizes, int n_in,
                              void* d_out, int out_size, void* d_ws, size_t ws_size,
                              hipStream_t stream) {
    const float* feat     = (const float*)d_in[0];
    const float* W_src    = (const float*)d_in[1];
    const float* b_src    = (const float*)d_in[2];
    const float* W_dst    = (const float*)d_in[3];
    const float* b_dst    = (const float*)d_in[4];
    const int*   user_ids = (const int*)d_in[5];
    const int*   item_ids = (const int*)d_in[6];
    const int*   edge_src = (const int*)d_in[7];
    const int*   edge_dst = (const int*)d_in[8];
    float* out = (float*)d_out;

    // workspace layout (all f32): feat_src | feat_dst | alpha | ex | scalars{max, inv_sum}
    char* ws = (char*)d_ws;
    float* feat_src = (float*)ws;  ws += (size_t)N_USERS * OUT_DIM * sizeof(float);
    float* feat_dst = (float*)ws;  ws += (size_t)N_ITEMS * OUT_DIM * sizeof(float);
    float* alpha    = (float*)ws;  ws += (size_t)N_EDGES * sizeof(float);
    float* ex       = (float*)ws;  ws += (size_t)N_EDGES * sizeof(float);
    float* scal     = (float*)ws;

    // 1. zero the (poisoned) output
    const int n4 = (N_NODES * OUT_DIM) / 4;
    k_zero<<<(n4 + 255) / 256, 256, 0, stream>>>(out, n4);

    // 2. edge logits (8 waves per 256-thread block)
    k_alpha<<<(N_EDGES + 7) / 8, 256, 0, stream>>>(feat, user_ids, item_ids,
                                                   edge_src, edge_dst, alpha);

    // 3/4. deterministic global softmax normalizers
    k_max<<<1, 1024, 0, stream>>>(alpha, scal);
    k_expsum<<<1, 1024, 0, stream>>>(alpha, ex, scal);

    // 5. WMMA GEMMs: relu(gather(feat)[M,256] @ W^T + b)
    dim3 gU((N_USERS / 16 + 3) / 4, OUT_DIM / 16);   // 782 x 16 blocks, 4 waves each
    k_gemm_relu<<<gU, 128, 0, stream>>>(feat, user_ids, W_src, b_src, feat_src, N_USERS);
    dim3 gI((N_ITEMS / 16 + 3) / 4, OUT_DIM / 16);   // 313 x 16
    k_gemm_relu<<<gI, 128, 0, stream>>>(feat, item_ids, W_dst, b_dst, feat_dst, N_ITEMS);

    // 6. alpha-weighted scatter in both directions
    k_scatter<<<N_EDGES, 256, 0, stream>>>(feat_src, feat_dst, edge_src, edge_dst,
                                           ex, scal, out);
}
